// Seq2SeqLSTM_72533407694922
// MI455X (gfx1250) — matrix operand order repaired and checked by reading, not compiled, not run on hardware
//
#include <hip/hip_runtime.h>
#include <math.h>
#include <stdint.h>

// ---------------------------------------------------------------------------
// Seq2Seq LSTM for MI455X (gfx1250, wave32, WMMA, TDM).
// bf16 WMMA (f32 accumulate) for every GEMM; persistent time-loop LSTM kernel
// with device-wide atomic barrier per step; batched logits GEMM with
// TDM (tensor_load_to_lds) double-buffered A staging.
// ---------------------------------------------------------------------------

typedef __attribute__((ext_vector_type(16))) __bf16    v16bf;
typedef __attribute__((ext_vector_type(8)))  __bf16    v8bf;
typedef __attribute__((ext_vector_type(8)))  float     v8f;
typedef __attribute__((ext_vector_type(4)))  unsigned  v4u;
typedef __attribute__((ext_vector_type(8)))  int       v8i;
typedef __attribute__((ext_vector_type(4)))  int       v4i;

#define CB  16      // batch (== WMMA M)
#define CH  1024    // hidden
#define CG  4096    // 4*H gate width

__device__ __forceinline__ v8f wmma_bf16(v16bf a, v16bf b, v8f c) {
  return __builtin_amdgcn_wmma_f32_16x16x32_bf16(false, a, false, b,
                                                 (short)0, c, false, false);
}

// A-fragment (16x32 bf16): lane holds row m = lane&15; elements [0..15] from
// K = kh*16 + 0..15  (kh = lane>>4).
// p must already point at (row m, k0 + kh*16).
__device__ __forceinline__ v16bf ldA16x32(const __bf16* p) {
  v8bf lo = *(const v8bf*)(p);
  v8bf hi = *(const v8bf*)(p + 8);
  v16bf r;
#pragma unroll
  for (int i = 0; i < 8; ++i) { r[i] = lo[i]; r[i + 8] = hi[i]; }
  return r;
}

__device__ __forceinline__ float sigmoidf_(float x) {
  return 1.0f / (1.0f + __expf(-x));
}

// Device-wide barrier: one fresh counter slot per (launch, timestep).
__device__ __forceinline__ void grid_barrier(unsigned* ctr, unsigned nwg) {
  __syncthreads();
  if (threadIdx.x == 0) {
    __threadfence();
    atomicAdd(ctr, 1u);
    while (atomicAdd(ctr, 0u) < nwg) { __builtin_amdgcn_s_sleep(1); }
    __threadfence();
  }
  __syncthreads();
}

// ---------------------------------------------------------------------------
// Tensor Data Mover: load a 2D tile (tile_dim0 = 32 K-elems contiguous,
// tile_dim1 = 64 rows, row stride = K elements) of bf16 data into LDS.
// D# bitfields per CDNA5 ISA ch.8 (groups 2/3 zero => 2D tensor).
// ---------------------------------------------------------------------------
__device__ __forceinline__ void tdm_load_a_tile(unsigned lds_off,
                                                const __bf16* gptr,
                                                int K, int Mtot) {
  unsigned long long ga = (unsigned long long)(uintptr_t)gptr;
  v4u g0;
  g0[0] = 1u;                                   // count = 1 valid descriptor
  g0[1] = lds_off;                              // lds_addr (bytes)
  g0[2] = (unsigned)(ga & 0xffffffffu);         // global_addr[31:0]
  g0[3] = (unsigned)((ga >> 32) & 0x01ffffffu)  // global_addr[56:32]
          | 0x80000000u;                        // type = 2 ("image")
  v8i g1;
  g1[0] = 0x00010000;                           // data_size = 1 (2 bytes)
  g1[1] = (int)((K & 0xffff) << 16);            // tensor_dim0 low16
  g1[2] = (int)(((K >> 16) & 0xffff)            // tensor_dim0 high16
          | ((Mtot & 0xffff) << 16));           // tensor_dim1 low16
  g1[3] = (int)(((Mtot >> 16) & 0xffff)         // tensor_dim1 high16
          | (32 << 16));                        // tile_dim0 = 32
  g1[4] = 64;                                   // tile_dim1 = 64, tile_dim2 = 0
  g1[5] = K;                                    // tensor_dim0_stride[31:0]
  g1[6] = 0;                                    // stride high / dim1_stride lo
  g1[7] = 0;
  v4i zz = {0, 0, 0, 0};
#if defined(__clang_major__) && (__clang_major__ >= 23)
  v8i zz8 = {0, 0, 0, 0, 0, 0, 0, 0};
  __builtin_amdgcn_tensor_load_to_lds(g0, g1, zz, zz, zz8, 0);
#else
  __builtin_amdgcn_tensor_load_to_lds(g0, g1, zz, zz, 0);
#endif
}

// ---------------------------------------------------------------------------
// Elementwise helpers
// ---------------------------------------------------------------------------
__global__ __launch_bounds__(256) void zero_u32_kernel(unsigned* p, long n) {
  long i = (long)blockIdx.x * 256 + threadIdx.x;
  long st = (long)gridDim.x * 256;
  for (; i < n; i += st) p[i] = 0u;
}

__global__ __launch_bounds__(256) void cvt_bf16_kernel(
    const float* __restrict__ s, __bf16* __restrict__ d, long n) {
  long i = (long)blockIdx.x * 256 + threadIdx.x;
  long st = (long)gridDim.x * 256;
  for (; i < n; i += st) d[i] = (__bf16)s[i];
}

// Embedding gather with padding_idx = 0 (row 0 -> zeros), fp32 -> bf16.
__global__ __launch_bounds__(256) void gather_emb_kernel(
    const int* __restrict__ tok, const float* __restrict__ emb,
    __bf16* __restrict__ out, int E) {
  int ti = blockIdx.x;
  int t = tok[ti];
  const float* row = emb + (long)t * E;
  __bf16* o = out + (long)ti * E;
  for (int e = threadIdx.x; e < E; e += 256)
    o[e] = (t == 0) ? (__bf16)0.0f : (__bf16)row[e];
}

// tanh(hc @ W.T + b): rows = 2*B = 32, cols = H, K = 2H. Tiny -> scalar FMA.
__global__ __launch_bounds__(256) void proj_kernel(
    const float* __restrict__ hc, const float* __restrict__ W,
    const float* __restrict__ bias, __bf16* __restrict__ out_bf,
    float* __restrict__ out_f) {
  int idx = blockIdx.x * 256 + threadIdx.x;     // 0 .. 32*1024-1
  int r = idx >> 10, c = idx & 1023;
  const float* a = hc + (long)r * (2 * CH);
  const float* w = W + (long)c * (2 * CH);
  float s = bias[c];
  for (int k = 0; k < 2 * CH; ++k) s = fmaf(a[k], w[k], s);
  s = tanhf(s);
  if (out_bf) out_bf[idx] = (__bf16)s;
  if (out_f)  out_f[idx] = s;
}

// ---------------------------------------------------------------------------
// Persistent LSTM layer. Grid = 32 WGs x 128 threads (4 waves).
// WG owns 32 hidden units; wave w owns gate w (i,f,g,o) -> 2 WMMA N-tiles.
// Per step: gates = x_t @ Wih.T + h_{t-1} @ Whh.T + bih + bhh (WMMA bf16),
// then cross-wave exchange in LDS, pointwise c/h update, grid barrier.
// ---------------------------------------------------------------------------
__global__ __launch_bounds__(128) void lstm_layer_kernel(
    const __bf16* __restrict__ x, long x_bs, long x_ts, int in_dim,
    const __bf16* __restrict__ Wih,   // (4H, in_dim) bf16
    const __bf16* __restrict__ Whh,   // (4H, H) bf16
    const float* __restrict__ bih, const float* __restrict__ bhh,
    __bf16* __restrict__ hseq, long h_bs, long h_ts,  // h_t out + recurrence in
    const __bf16* __restrict__ h_init,                // (B,H) or null
    float* __restrict__ c_state,                      // (B,H) pre-initialized
    float* __restrict__ final_h, float* __restrict__ final_c, long fin_rs,
    int T, int reverse, unsigned* __restrict__ barrier_base) {
  const int hid0 = blockIdx.x * 32;
  const int tid  = threadIdx.x;
  const int gate = tid >> 5;            // wave index == gate index
  const int lane = tid & 31;
  const int nrow = lane & 15;
  const int kh   = lane >> 4;
  const int col0 = gate * CH + hid0;    // first column of this wave's tile 0

  __shared__ float gbuf[4][CB][32];     // [gate][batch][hidden-offset]

  const float bias0 = bih[col0 + nrow] + bhh[col0 + nrow];
  const float bias1 = bih[col0 + 16 + nrow] + bhh[col0 + 16 + nrow];
  const __bf16* wih0 = Wih + (long)(col0 + nrow) * in_dim + kh * 16;
  const __bf16* wih1 = Wih + (long)(col0 + 16 + nrow) * in_dim + kh * 16;
  const __bf16* whh0 = Whh + (long)(col0 + nrow) * CH + kh * 16;
  const __bf16* whh1 = Whh + (long)(col0 + 16 + nrow) * CH + kh * 16;

  for (int s = 0; s < T; ++s) {
    const int t = reverse ? (T - 1 - s) : s;
    v8f acc0 = {}, acc1 = {};

    // ---- gates += x_t @ Wih.T ----
    const __bf16* xa = x + (long)nrow * x_bs + (long)t * x_ts + kh * 16;
    for (int k0 = 0; k0 < in_dim; k0 += 32) {
      v16bf a = ldA16x32(xa + k0);
      acc0 = wmma_bf16(a, *(const v16bf*)(wih0 + k0), acc0);
      acc1 = wmma_bf16(a, *(const v16bf*)(wih1 + k0), acc1);
    }

    // ---- gates += h_{t-1} @ Whh.T ----
    const __bf16* hp = nullptr;
    long hp_rs = 0;
    if (s == 0) { hp = h_init; hp_rs = CH; }
    else { int pt = reverse ? (t + 1) : (t - 1); hp = hseq + (long)pt * h_ts; hp_rs = h_bs; }
    if (hp) {
      const __bf16* ha = hp + (long)nrow * hp_rs + kh * 16;
      for (int k0 = 0; k0 < CH; k0 += 32) {
        v16bf a = ldA16x32(ha + k0);
        acc0 = wmma_bf16(a, *(const v16bf*)(whh0 + k0), acc0);
        acc1 = wmma_bf16(a, *(const v16bf*)(whh1 + k0), acc1);
      }
    }

    // ---- stash gates to LDS (D layout: m = kh*8+e, n = nrow) ----
#pragma unroll
    for (int e = 0; e < 8; ++e) {
      gbuf[gate][kh * 8 + e][nrow]      = acc0[e] + bias0;
      gbuf[gate][kh * 8 + e][16 + nrow] = acc1[e] + bias1;
    }
    __syncthreads();

    // ---- pointwise cell update: 16 batches x 32 hidden units ----
    for (int p = tid; p < CB * 32; p += 128) {
      int b = p >> 5, j = p & 31;
      float iv = sigmoidf_(gbuf[0][b][j]);
      float fv = sigmoidf_(gbuf[1][b][j]);
      float gv = tanhf(gbuf[2][b][j]);
      float ov = sigmoidf_(gbuf[3][b][j]);
      float* cp = c_state + (long)b * CH + hid0 + j;
      float cv = fv * (*cp) + iv * gv;
      *cp = cv;
      float hv = ov * tanhf(cv);
      hseq[(long)b * h_bs + (long)t * h_ts + hid0 + j] = (__bf16)hv;
      if (s == T - 1 && final_h) {
        final_h[(long)b * fin_rs + hid0 + j] = hv;
        final_c[(long)b * fin_rs + hid0 + j] = cv;
      }
    }
    grid_barrier(barrier_base + s, gridDim.x);   // h_t visible device-wide
  }
}

// ---------------------------------------------------------------------------
// Logits GEMM: out(M=1024, N=32000) = A(1024,1024) @ W.T + bias.
// WG = 256 thr (8 waves), tile 64M x 128N; wave = (msub 0..3, nsub 0..1)
// computes 16M x 64N via 4 accumulators. A tile staged into LDS by the
// Tensor Data Mover (double buffered: TDM block k+1 overlaps WMMA block k).
// ---------------------------------------------------------------------------
__global__ __launch_bounds__(256) void logits_gemm_kernel(
    const __bf16* __restrict__ A, const __bf16* __restrict__ W,
    const float* __restrict__ bias, float* __restrict__ out,
    int M, int N, int K) {
  __shared__ __align__(16) __bf16 lds_a[2][64][32];
  const int tid  = threadIdx.x;
  const int wv   = tid >> 5, lane = tid & 31;
  const int nrow = lane & 15, kh = lane >> 4;
  const int msub = wv & 3, nsub = wv >> 2;
  const int m0 = blockIdx.y * 64;
  const int n0 = blockIdx.x * 128 + nsub * 64;

  v8f z = {};
  v8f acc[4] = {z, z, z, z};
  const __bf16* wp[4];
#pragma unroll
  for (int nt = 0; nt < 4; ++nt)
    wp[nt] = W + (long)(n0 + nt * 16 + nrow) * K + kh * 16;
  const __bf16* abase = A + (long)m0 * K;
  const unsigned lds0 = (unsigned)(uintptr_t)(&lds_a[0][0][0]);
  const unsigned lds1 = (unsigned)(uintptr_t)(&lds_a[1][0][0]);

  // Prologue: TDM-load block 0 into buffer 0.
  if (wv == 0) {
    tdm_load_a_tile(lds0, abase, K, M);
    __builtin_amdgcn_s_wait_tensorcnt(0);
  }
  __syncthreads();

  int pb = 0;
  for (int k0 = 0; k0 < K; k0 += 32) {
    // Kick off the DMA for the next K-block while this one is consumed.
    if (wv == 0 && (k0 + 32) < K)
      tdm_load_a_tile(pb ? lds0 : lds1, abase + k0 + 32, K, M);
    if (k0 + 32 < K) {
#pragma unroll
      for (int nt = 0; nt < 4; ++nt)
        __builtin_prefetch((const void*)(wp[nt] + k0 + 32), 0, 1);
    }
    v16bf a = ldA16x32(&lds_a[pb][msub * 16 + nrow][kh * 16]);
#pragma unroll
    for (int nt = 0; nt < 4; ++nt)
      acc[nt] = wmma_bf16(a, *(const v16bf*)(wp[nt] + k0), acc[nt]);
    if (wv == 0 && (k0 + 32) < K)
      __builtin_amdgcn_s_wait_tensorcnt(0);
    __syncthreads();                     // publish DMA'd buffer to all waves
    pb ^= 1;
  }

#pragma unroll
  for (int nt = 0; nt < 4; ++nt) {
    int col = n0 + nt * 16 + nrow;
    float bv = bias[col];
#pragma unroll
    for (int e = 0; e < 8; ++e) {
      int m = m0 + msub * 16 + kh * 8 + e;
      out[(long)m * N + col] = acc[nt][e] + bv;
    }
  }
}

// ---------------------------------------------------------------------------
// Host orchestration
// ---------------------------------------------------------------------------
extern "C" void kernel_launch(void* const* d_in, const int* in_sizes, int n_in,
                              void* d_out, int out_size, void* d_ws,
                              size_t ws_size, hipStream_t stream) {
  (void)in_sizes; (void)n_in; (void)out_size; (void)ws_size;
  const int B = 16, S = 128, T = 64, E = 512, H = 1024, V = 32000, G = 4 * H;

  const int*   src    = (const int*)d_in[0];
  const int*   tgt    = (const int*)d_in[1];
  const float* emb    = (const float*)d_in[2];
  const float* eWih0  = (const float*)d_in[3];
  const float* eWhh0  = (const float*)d_in[4];
  const float* ebih0  = (const float*)d_in[5];
  const float* ebhh0  = (const float*)d_in[6];
  const float* eWih1  = (const float*)d_in[7];
  const float* eWhh1  = (const float*)d_in[8];
  const float* ebih1  = (const float*)d_in[9];
  const float* ebhh1  = (const float*)d_in[10];
  const float* dWih0f = (const float*)d_in[11];
  const float* dWhh0f = (const float*)d_in[12];
  const float* dbih0  = (const float*)d_in[13];
  const float* dbhh0  = (const float*)d_in[14];
  const float* dWih1f = (const float*)d_in[15];
  const float* dWhh1f = (const float*)d_in[16];
  const float* dbih1  = (const float*)d_in[17];
  const float* dbhh1  = (const float*)d_in[18];
  const float* projhW = (const float*)d_in[19];
  const float* projhB = (const float*)d_in[20];
  const float* projcW = (const float*)d_in[21];
  const float* projcB = (const float*)d_in[22];
  const float* fcWf   = (const float*)d_in[23];
  const float* fcb    = (const float*)d_in[24];
  float* out = (float*)d_out;

  size_t off = 0;
  auto take = [&](size_t bytes) -> char* {
    char* p = (char*)d_ws + off;
    off = (off + bytes + 255) & ~(size_t)255;
    return p;
  };
  auto take_bf = [&](long n) { return (__bf16*)take((size_t)n * 2); };
  auto take_f  = [&](long n) { return (float*)take((size_t)n * 4); };

  // bf16 weight arena
  __bf16* bWih0  = take_bf(2L * G * E);
  __bf16* bWhh0  = take_bf(2L * G * H);
  __bf16* bWih1  = take_bf(2L * G * 2 * H);
  __bf16* bWhh1  = take_bf(2L * G * H);
  __bf16* bdWih0 = take_bf((long)G * E);
  __bf16* bdWhh0 = take_bf((long)G * H);
  __bf16* bdWih1 = take_bf((long)G * H);
  __bf16* bdWhh1 = take_bf((long)G * H);
  __bf16* bfcW   = take_bf((long)V * H);
  // activations
  __bf16* embS   = take_bf((long)B * S * E);
  __bf16* embT   = take_bf((long)B * T * E);
  __bf16* y0     = take_bf((long)B * S * 2 * H);   // enc L0 fwd|bwd concat
  __bf16* h1f    = take_bf((long)B * S * H);
  __bf16* h1b    = take_bf((long)B * S * H);
  __bf16* dh0seq = take_bf((long)B * T * H);
  __bf16* dh1seq = take_bf((long)B * T * H);       // rows b*T+t for GEMM
  __bf16* hinit  = take_bf(2L * B * H);
  float*  cste   = take_f(4L * B * H);             // enc cell states (4 dirs)
  float*  cdec   = take_f(2L * B * H);             // dec cell states (2 layers)
  float*  hcat   = take_f(2L * B * 2 * H);         // (2,B,2H) final h
  float*  ccat   = take_f(2L * B * 2 * H);
  unsigned* bars = (unsigned*)take(4096 * 4);

  // 0) zero cell states + barrier counters (deterministic each call)
  zero_u32_kernel<<<64, 256, 0, stream>>>((unsigned*)cste, 4L * B * H);
  zero_u32_kernel<<<4, 256, 0, stream>>>(bars, 4096L);

  // 1) fp32 -> bf16 weight conversion (one streaming pass)
  cvt_bf16_kernel<<<2048, 256, 0, stream>>>(eWih0,  bWih0,  2L * G * E);
  cvt_bf16_kernel<<<2048, 256, 0, stream>>>(eWhh0,  bWhh0,  2L * G * H);
  cvt_bf16_kernel<<<2048, 256, 0, stream>>>(eWih1,  bWih1,  2L * G * 2 * H);
  cvt_bf16_kernel<<<2048, 256, 0, stream>>>(eWhh1,  bWhh1,  2L * G * H);
  cvt_bf16_kernel<<<2048, 256, 0, stream>>>(dWih0f, bdWih0, (long)G * E);
  cvt_bf16_kernel<<<2048, 256, 0, stream>>>(dWhh0f, bdWhh0, (long)G * H);
  cvt_bf16_kernel<<<2048, 256, 0, stream>>>(dWih1f, bdWih1, (long)G * H);
  cvt_bf16_kernel<<<2048, 256, 0, stream>>>(dWhh1f, bdWhh1, (long)G * H);
  cvt_bf16_kernel<<<2048, 256, 0, stream>>>(fcWf,   bfcW,   (long)V * H);

  // 2) embedding gathers (padding_idx = 0)
  gather_emb_kernel<<<B * S, 256, 0, stream>>>(src, emb, embS, E);
  gather_emb_kernel<<<B * T, 256, 0, stream>>>(tgt, emb, embT, E);

  // 3) encoder layer 0 (bidirectional) -> y0
  lstm_layer_kernel<<<32, 128, 0, stream>>>(
      embS, (long)S * E, (long)E, E, bWih0, bWhh0, ebih0, ebhh0,
      y0, (long)S * 2 * H, (long)2 * H, nullptr, cste,
      hcat, ccat, (long)2 * H, S, 0, bars + 0);
  lstm_layer_kernel<<<32, 128, 0, stream>>>(
      embS, (long)S * E, (long)E, E, bWih0 + (long)G * E, bWhh0 + (long)G * H,
      ebih0 + G, ebhh0 + G,
      y0 + H, (long)S * 2 * H, (long)2 * H, nullptr, cste + (long)B * H,
      hcat + H, ccat + H, (long)2 * H, S, 1, bars + 128);

  // 4) encoder layer 1 (bidirectional), inputs y0 (2H)
  lstm_layer_kernel<<<32, 128, 0, stream>>>(
      y0, (long)S * 2 * H, (long)2 * H, 2 * H, bWih1, bWhh1, ebih1, ebhh1,
      h1f, (long)S * H, (long)H, nullptr, cste + 2L * B * H,
      hcat + 2L * B * H, ccat + 2L * B * H, (long)2 * H, S, 0, bars + 256);
  lstm_layer_kernel<<<32, 128, 0, stream>>>(
      y0, (long)S * 2 * H, (long)2 * H, 2 * H, bWih1 + (long)G * 2 * H,
      bWhh1 + (long)G * H, ebih1 + G, ebhh1 + G,
      h1b, (long)S * H, (long)H, nullptr, cste + 3L * B * H,
      hcat + 2L * B * H + H, ccat + 2L * B * H + H, (long)2 * H, S, 1,
      bars + 384);

  // 5) bridge projections: hidden -> bf16 decoder h_init, cell -> f32 c state
  proj_kernel<<<128, 256, 0, stream>>>(hcat, projhW, projhB, hinit, nullptr);
  proj_kernel<<<128, 256, 0, stream>>>(ccat, projcW, projcB, nullptr, cdec);

  // 6) decoder: cell0 over all T (teacher forcing), then cell1 over all T
  lstm_layer_kernel<<<32, 128, 0, stream>>>(
      embT, (long)T * E, (long)E, E, bdWih0, bdWhh0, dbih0, dbhh0,
      dh0seq, (long)T * H, (long)H, hinit, cdec,
      nullptr, nullptr, 0, T, 0, bars + 512);
  lstm_layer_kernel<<<32, 128, 0, stream>>>(
      dh0seq, (long)T * H, (long)H, H, bdWih1, bdWhh1, dbih1, dbhh1,
      dh1seq, (long)T * H, (long)H, hinit + (long)B * H, cdec + (long)B * H,
      nullptr, nullptr, 0, T, 0, bars + 640);

  // 7) batched logits GEMM: (B*T, H) @ fc_W.T + fc_b -> (B, T, V)
  logits_gemm_kernel<<<dim3(V / 128, (B * T) / 64), 256, 0, stream>>>(
      dh1seq, bfcW, fcb, out, B * T, V, H);
}
